// MultiHeadAttention_48395691491372
// MI455X (gfx1250) — compile-verified, matrix operand order checked by reading, not run on hardware
//
#include <hip/hip_runtime.h>
#include <hip/hip_bf16.h>

typedef __attribute__((ext_vector_type(16))) _Float16 v16h;
typedef __attribute__((ext_vector_type(8)))  _Float16 v8h;
typedef __attribute__((ext_vector_type(8)))  float    v8f;

#define B_  2
#define S_  2048
#define D_  1024
#define H_  16
#define HD_ 64
#define NB_ 16      // S_/128
#define TD3_ 3072   // 3*D_

__device__ __forceinline__ v16h mkfrag(v8h lo, v8h hi) {
    v16h r;
#pragma unroll
    for (int i = 0; i < 8; ++i) { r[i] = lo[i]; r[8 + i] = hi[i]; }
    return r;
}

__device__ __forceinline__ v8f wmma16(v16h a, v16h b, v8f c) {
    return __builtin_amdgcn_wmma_f32_16x16x32_f16(false, a, false, b, (short)0, c, false, false);
}

// CDNA5 async global->LDS copy (ASYNCcnt-tracked, bypasses VGPRs)
__device__ __forceinline__ void async_copy_b128(unsigned int lds_off, unsigned long long gaddr) {
    asm volatile("global_load_async_to_lds_b128 %0, %1, off"
                 :: "v"(lds_off), "v"(gaddr) : "memory");
}
__device__ __forceinline__ void wait_asynccnt0() {
    asm volatile("s_wait_asynccnt 0" ::: "memory");
}
__device__ __forceinline__ unsigned int lds_off_u32(const void* p) {
    return (unsigned int)(unsigned long long)p;   // addr[31:0] = LDS offset
}

// ---------------------------------------------------------------- convert
__global__ void cvt_f32_to_f16(const float* __restrict__ in, _Float16* __restrict__ out, int n) {
    int i = blockIdx.x * blockDim.x + threadIdx.x;
    if (i < n) out[i] = (_Float16)in[i];
}

// ---------------------------------------------------------------- GEMM
// C[M,N] = A[M,K] * B[K,N] ; A,B f16 row-major, f32 accumulate.
// Block tile 128x128, 256 threads (8 waves), wave tile 32x64 (2x4 wmma tiles).
// Double-buffered LDS: async A-copy + B staging for tile k+1 overlap WMMAs on tile k.
template <bool F32OUT>
__global__ void __launch_bounds__(256, 2)
gemm_wmma_f16(const _Float16* __restrict__ A,
              const _Float16* __restrict__ Bm,
              _Float16* __restrict__ Cf16,
              float* __restrict__ Cf32,
              const float* __restrict__ bias,
              int M, int N, int K) {
    __shared__ alignas(16) _Float16 Alds[2][128][40];   // [buf][m][k], 80B stride
    __shared__ alignas(16) _Float16 Blds[2][128][40];   // [buf][n][k] (transposed)

    const int t    = threadIdx.x;
    const int lane = t & 31;
    const int wave = t >> 5;
    const int bm   = blockIdx.y * 128;
    const int bn   = blockIdx.x * 128;
    const int wm   = (wave >> 1) * 32;
    const int wn   = (wave & 1) * 64;
    const int fr   = lane & 15;
    const int kb8  = (lane >> 4) * 8;

    v8f acc[2][4];
#pragma unroll
    for (int i = 0; i < 2; ++i)
#pragma unroll
        for (int j = 0; j < 4; ++j) acc[i][j] = (v8f)(0.0f);

    const unsigned int albase[2] = { lds_off_u32(&Alds[0][0][0]), lds_off_u32(&Alds[1][0][0]) };
    const int ar = t >> 2;          // 0..63?  (A staging: see id mapping below)
    const int kr = t >> 3;          // 0..31  (B staging row)
    const int bc = (t & 7) * 16;    // 0..112 (B staging col chunk)
    (void)ar;

    // ---- prologue: stage tile kt=0 into buffer 0
    {
#pragma unroll
        for (int c = 0; c < 2; ++c) {
            const int id  = c * 256 + t;
            const int row = id >> 2;          // 0..127
            const int ch  = id & 3;           // 0..3 (x8 halves)
            async_copy_b128(albase[0] + (unsigned)(row * 80 + ch * 16),
                            (unsigned long long)(A + (size_t)(bm + row) * K + ch * 8));
        }
        const _Float16* bg = Bm + (size_t)kr * N + bn + bc;
        v8h b0 = *(const v8h*)(bg);
        v8h b1 = *(const v8h*)(bg + 8);
#pragma unroll
        for (int i = 0; i < 8; ++i) {
            Blds[0][bc + i][kr]     = b0[i];
            Blds[0][bc + 8 + i][kr] = b1[i];
        }
        wait_asynccnt0();
        __syncthreads();
    }

    int cur = 0;
    for (int kt = 0; kt < K; kt += 32) {
        const int  nxt       = cur ^ 1;
        const bool have_next = (kt + 32) < K;

        v8h nb0, nb1;
        if (have_next) {
            // issue next A tile async (no VGPR round-trip) and next B tile global loads
#pragma unroll
            for (int c = 0; c < 2; ++c) {
                const int id  = c * 256 + t;
                const int row = id >> 2;
                const int ch  = id & 3;
                async_copy_b128(albase[nxt] + (unsigned)(row * 80 + ch * 16),
                                (unsigned long long)(A + (size_t)(bm + row) * K + kt + 32 + ch * 8));
            }
            const _Float16* bg = Bm + (size_t)(kt + 32 + kr) * N + bn + bc;
            nb0 = *(const v8h*)(bg);
            nb1 = *(const v8h*)(bg + 8);
        }

        // ---- compute from buffer cur
        v16h af[2];
#pragma unroll
        for (int i = 0; i < 2; ++i) {
            const _Float16* p = &Alds[cur][wm + i * 16 + fr][0];
            af[i] = mkfrag(*(const v8h*)(p + kb8), *(const v8h*)(p + 16 + kb8));
        }
        v16h bf[4];
#pragma unroll
        for (int j = 0; j < 4; ++j) {
            const _Float16* p = &Blds[cur][wn + j * 16 + fr][0];
            bf[j] = mkfrag(*(const v8h*)(p + kb8), *(const v8h*)(p + 16 + kb8));
        }
#pragma unroll
        for (int i = 0; i < 2; ++i)
#pragma unroll
            for (int j = 0; j < 4; ++j)
                acc[i][j] = wmma16(af[i], bf[j], acc[i][j]);

        if (have_next) {
            // scatter next B tile (writes buf nxt; concurrent reads are from buf cur)
#pragma unroll
            for (int i = 0; i < 8; ++i) {
                Blds[nxt][bc + i][kr]     = nb0[i];
                Blds[nxt][bc + 8 + i][kr] = nb1[i];
            }
            wait_asynccnt0();
        }
        __syncthreads();
        cur = nxt;
    }

    const int rhi = (lane >> 4) * 8;
#pragma unroll
    for (int i = 0; i < 2; ++i) {
#pragma unroll
        for (int j = 0; j < 4; ++j) {
            const int col = bn + wn + j * 16 + fr;
#pragma unroll
            for (int e = 0; e < 8; ++e) {
                const int row = bm + wm + i * 16 + rhi + e;
                float v = acc[i][j][e];
                if (F32OUT) {
                    Cf32[(size_t)row * N + col] = v + bias[col];
                } else {
                    Cf16[(size_t)row * N + col] = (_Float16)v;
                }
            }
        }
    }
}

// ---------------------------------------------------------------- attention
// qkv: [B*S][3072] f16 (q | k | v, each head 64 contiguous)
// out: [B*S][1024] f16
// one workgroup (256 thr, 8 waves) per (qblock, head, batch); wave owns 16 q rows.
__global__ void __launch_bounds__(256, 2)
attn_flash_wmma(const _Float16* __restrict__ qkv,
                const unsigned char* __restrict__ mask,
                _Float16* __restrict__ outh) {
    const int qb = blockIdx.x;   // 0..15
    const int h  = blockIdx.y;   // 0..15
    const int b  = blockIdx.z;   // 0..1

    // Klds (QK^T phase) and Plds (PV phase) are never live together -> union.
    __shared__ alignas(16) _Float16 UnionBuf[8 * 16 * 136];   // 34,816 B
    __shared__ alignas(16) _Float16 Vt[64][136];              // V^T: [hd][k]
    _Float16 (*Klds)[72]       = (_Float16(*)[72])UnionBuf;       // [k_row][hd], 144B stride
    _Float16 (*Plds)[16][136]  = (_Float16(*)[16][136])UnionBuf;  // per-wave P tile

    const int t    = threadIdx.x;
    const int lane = t & 31;
    const int wave = t >> 5;
    const int fr   = lane & 15;
    const int kb8  = (lane >> 4) * 8;
    const int rhi  = (lane >> 4) * 8;

    const size_t seq0 = (size_t)b * S_;
    const unsigned int klbase = lds_off_u32(&Klds[0][0]);

    // persistent Q fragments (16 rows x 64 hd = 2 k-steps)
    const _Float16* qbase = qkv + (seq0 + qb * 128 + wave * 16 + fr) * TD3_ + h * HD_;
    v16h qf[2];
#pragma unroll
    for (int ks = 0; ks < 2; ++ks)
        qf[ks] = mkfrag(*(const v8h*)(qbase + ks * 32 + kb8),
                        *(const v8h*)(qbase + ks * 32 + 16 + kb8));

    v8f oacc[4];
#pragma unroll
    for (int j = 0; j < 4; ++j) oacc[j] = (v8f)(0.0f);
    float rmax[8], rsum[8];
#pragma unroll
    for (int e = 0; e < 8; ++e) { rmax[e] = -__builtin_inff(); rsum[e] = 0.0f; }

    const float scale = 0.125f;  // 1/sqrt(64)

    for (int kb = 0; kb < NB_; ++kb) {
        if (mask[qb * NB_ + kb]) continue;   // uniform across workgroup

        // ---- stage K block (128x64) raw into Klds via async copy: 1024 chunks
        {
            const _Float16* kg = qkv + (seq0 + kb * 128) * TD3_ + D_ + h * HD_;
#pragma unroll
            for (int c = 0; c < 4; ++c) {
                const int id  = c * 256 + t;
                const int row = id >> 3;     // 0..127
                const int ch  = id & 7;      // 0..7 (x8 halves)
                async_copy_b128(klbase + (unsigned)(row * 144 + ch * 16),
                                (unsigned long long)(kg + (size_t)row * TD3_ + ch * 8));
            }
        }
        // ---- stage V block (128x64) transposed into Vt
        {
            const int kr = t >> 1;           // 0..127
            const int dc = (t & 1) * 32;     // 0 / 32
            const _Float16* vg = qkv + (seq0 + kb * 128 + kr) * TD3_ + 2 * D_ + h * HD_ + dc;
#pragma unroll
            for (int c = 0; c < 4; ++c) {
                v8h v = *(const v8h*)(vg + c * 8);
#pragma unroll
                for (int i = 0; i < 8; ++i) Vt[dc + c * 8 + i][kr] = v[i];
            }
        }
        wait_asynccnt0();
        __syncthreads();

        // ---- S = Q K^T : 8 tiles of 16 cols, K fragments from LDS
        v8f st[8];
#pragma unroll
        for (int nt = 0; nt < 8; ++nt) {
            v8f c = (v8f)(0.0f);
#pragma unroll
            for (int ks = 0; ks < 2; ++ks) {
                const _Float16* kp = &Klds[nt * 16 + fr][ks * 32 + kb8];
                v16h kf = mkfrag(*(const v8h*)(kp), *(const v8h*)(kp + 16));
                c = wmma16(qf[ks], kf, c);
            }
            st[nt] = c;
        }
        __syncthreads();   // all K reads done before P overwrites UnionBuf

        // ---- online softmax (row r split across 16 lanes)
        float bmax[8];
#pragma unroll
        for (int e = 0; e < 8; ++e) bmax[e] = -__builtin_inff();
#pragma unroll
        for (int nt = 0; nt < 8; ++nt)
#pragma unroll
            for (int e = 0; e < 8; ++e) {
                st[nt][e] *= scale;
                bmax[e] = fmaxf(bmax[e], st[nt][e]);
            }
#pragma unroll
        for (int m = 1; m < 16; m <<= 1)
#pragma unroll
            for (int e = 0; e < 8; ++e)
                bmax[e] = fmaxf(bmax[e], __shfl_xor(bmax[e], m, 32));

        float corr[8];
#pragma unroll
        for (int e = 0; e < 8; ++e) {
            float nm = fmaxf(rmax[e], bmax[e]);
            corr[e] = __expf(rmax[e] - nm);
            rmax[e] = nm;
        }
        float bsum[8];
#pragma unroll
        for (int e = 0; e < 8; ++e) bsum[e] = 0.0f;
#pragma unroll
        for (int nt = 0; nt < 8; ++nt)
#pragma unroll
            for (int e = 0; e < 8; ++e) {
                float p = __expf(st[nt][e] - rmax[e]);
                st[nt][e] = p;
                bsum[e] += p;
            }
#pragma unroll
        for (int m = 1; m < 16; m <<= 1)
#pragma unroll
            for (int e = 0; e < 8; ++e) bsum[e] += __shfl_xor(bsum[e], m, 32);
#pragma unroll
        for (int e = 0; e < 8; ++e) rsum[e] = rsum[e] * corr[e] + bsum[e];
#pragma unroll
        for (int j = 0; j < 4; ++j)
#pragma unroll
            for (int e = 0; e < 8; ++e) oacc[j][e] *= corr[e];

        // ---- stage P (C layout -> LDS) for re-read as A fragments
#pragma unroll
        for (int nt = 0; nt < 8; ++nt)
#pragma unroll
            for (int e = 0; e < 8; ++e)
                Plds[wave][rhi + e][nt * 16 + fr] = (_Float16)st[nt][e];
        __syncthreads();

        // ---- O += P V
#pragma unroll
        for (int ks = 0; ks < 4; ++ks) {
            const _Float16* pp = &Plds[wave][fr][ks * 32];
            v16h pf = mkfrag(*(const v8h*)(pp + kb8), *(const v8h*)(pp + 16 + kb8));
#pragma unroll
            for (int j = 0; j < 4; ++j) {
                const _Float16* vp = &Vt[j * 16 + fr][ks * 32];
                v16h vf = mkfrag(*(const v8h*)(vp + kb8), *(const v8h*)(vp + 16 + kb8));
                oacc[j] = wmma16(pf, vf, oacc[j]);
            }
        }
        __syncthreads();   // protect Vt/UnionBuf for next iteration
    }

    // ---- normalize + store f16
#pragma unroll
    for (int j = 0; j < 4; ++j)
#pragma unroll
        for (int e = 0; e < 8; ++e) {
            const size_t row = seq0 + qb * 128 + wave * 16 + rhi + e;
            outh[row * D_ + h * HD_ + j * 16 + fr] = (_Float16)(oacc[j][e] / rsum[e]);
        }
}

// ---------------------------------------------------------------- launch
extern "C" void kernel_launch(void* const* d_in, const int* in_sizes, int n_in,
                              void* d_out, int out_size, void* d_ws, size_t ws_size,
                              hipStream_t stream) {
    const float* x      = (const float*)d_in[0];          // [2,2048,1024]
    const float* w_qkv  = (const float*)d_in[1];          // [1024,3072]
    const float* w_proj = (const float*)d_in[2];          // [1024,1024]
    const float* b_proj = (const float*)d_in[3];          // [1024]
    const unsigned char* bmask = (const unsigned char*)d_in[4]; // [16,16] bool

    const int M = B_ * S_;          // 4096
    const int nX  = M * D_;         // 4,194,304
    const int nWQ = D_ * 3 * D_;    // 3,145,728
    const int nWP = D_ * D_;        // 1,048,576

    _Float16* xh   = (_Float16*)d_ws;
    _Float16* wqh  = xh  + (size_t)nX;
    _Float16* wph  = wqh + (size_t)nWQ;
    _Float16* qkvh = wph + (size_t)nWP;
    _Float16* aoh  = qkvh + (size_t)M * TD3_;

    // 1) f32 -> f16 conversions
    cvt_f32_to_f16<<<(nX  + 255) / 256, 256, 0, stream>>>(x,      xh,  nX);
    cvt_f32_to_f16<<<(nWQ + 255) / 256, 256, 0, stream>>>(w_qkv,  wqh, nWQ);
    cvt_f32_to_f16<<<(nWP + 255) / 256, 256, 0, stream>>>(w_proj, wph, nWP);

    // 2) qkv = x @ w_qkv    (4096x1024 @ 1024x3072)
    gemm_wmma_f16<false><<<dim3(3 * D_ / 128, M / 128), 256, 0, stream>>>(
        xh, wqh, qkvh, nullptr, nullptr, M, 3 * D_, D_);

    // 3) block-sparse flash attention
    attn_flash_wmma<<<dim3(NB_, H_, B_), 256, 0, stream>>>(qkvh, bmask, aoh);

    // 4) out = attn @ w_proj + b_proj  (f32 output)
    gemm_wmma_f16<true><<<dim3(D_ / 128, M / 128), 256, 0, stream>>>(
        aoh, wph, nullptr, (float*)d_out, b_proj, M, D_, D_);
}
